// AttentionToMotion_90348932038886
// MI455X (gfx1250) — compile-verified, matrix operand order checked by reading, not run on hardware
//
#include <hip/hip_runtime.h>
#include <math.h>

typedef __attribute__((ext_vector_type(16))) _Float16 v16h;
typedef __attribute__((ext_vector_type(8)))  _Float16 v8h;
typedef __attribute__((ext_vector_type(8)))  float    v8f;
typedef __attribute__((ext_vector_type(4)))  float    v4f;

#define DIM   256
#define NTOK  64
#define HEADS 8
#define HD    32
#define ATT_SCALE 0.17677669529663689f   // 1/sqrt(32)

union V16U { v16h v; v8h h[2]; };

__device__ __forceinline__ v8f wmma16(v16h a, v16h b, v8f c) {
  // D = A(16x32 f16) * B(32x16 f16) + C(16x16 f32)
  return __builtin_amdgcn_wmma_f32_16x16x32_f16(false, a, false, b, (short)0, c,
                                                false, false);
}

__device__ __forceinline__ v16h ld_frag_lds(const _Float16* base, int frag, int lane) {
  return *(const v16h*)(base + (frag * 32 + lane) * 16);
}
__device__ __forceinline__ v16h ld_frag_glb(const _Float16* __restrict__ base, int frag, int lane) {
  return *(const v16h*)(base + (frag * 32 + lane) * 16);
}

// -------------------------------------------------------------------------
// Weight prep: fp32 row-major (K x N) -> f16 fragment-swizzled.
// Fragment (kTile= k/32, nTile= n/16): lane = (n%16) + 16*g, half e, where
//   kk=k%32, g=(kk>>3)&1, e=((kk>>4)<<3)+(kk&7)
// ws halves layout: Wq [0,65536) Wkv [65536,196608) Wproj [196608,262144)
// -------------------------------------------------------------------------
__global__ __launch_bounds__(256) void prep_weights_kernel(
    const float* __restrict__ Wq, const float* __restrict__ Wkv,
    const float* __restrict__ Wproj, _Float16* __restrict__ ws)
{
  int idx = blockIdx.x * 256 + threadIdx.x;   // 0 .. 262143
  const float* src; _Float16* dst; int Nn, off;
  if (idx < 65536)       { src = Wq;    dst = ws;          Nn = 256; off = idx; }
  else if (idx < 196608) { src = Wkv;   dst = ws + 65536;  Nn = 512; off = idx - 65536; }
  else                   { src = Wproj; dst = ws + 196608; Nn = 256; off = idx - 196608; }
  int k = off / Nn, n = off - k * Nn;
  int kk = k & 31;
  int g = (kk >> 3) & 1;
  int e = ((kk >> 4) << 3) + (kk & 7);
  int frag = (k >> 5) * (Nn >> 4) + (n >> 4);
  dst[(frag * 32 + ((n & 15) + (g << 4))) * 16 + e] = (_Float16)src[(size_t)k * Nn + n];
}

// -------------------------------------------------------------------------
// Main fused kernel: one block per window (batch element), 8 waves.
// -------------------------------------------------------------------------
__global__ __launch_bounds__(256) void attn_motion_kernel(
    const float* __restrict__ gx1, const float* __restrict__ gx2,
    const _Float16* __restrict__ wsw, const float* __restrict__ bproj,
    const float* __restrict__ Wm1, const float* __restrict__ bm1,
    const float* __restrict__ Wm2, const float* __restrict__ bm2,
    float* __restrict__ out, int nB)
{
  __shared__ _Float16 sm[81920];                 // 160 KB
  _Float16* const smX1 = sm;                     // x1 frags, later xattn frags
  _Float16* const smX2 = sm + 16384;             // x2 frags, later P-stage+motion
  _Float16* const smQ  = sm + 32768;             // q  A-frags per (head, mTile)
  _Float16* const smK  = sm + 49152;             // kT B-frags per (head, tokTile)
  _Float16* const smV  = sm + 65536;             // v  B-frags per (head, kt, nt)
  _Float16* const smStage = smX2;                // 8 waves * 1024 halves
  float*    const smMotion = (float*)(smX2 + 8192); // [2][8][64] f32

  const _Float16* const gWq    = wsw;
  const _Float16* const gWkv   = wsw + 65536;
  const _Float16* const gWproj = wsw + 196608;

  const int tid  = threadIdx.x;
  const int lane = tid & 31;
  const int wave = tid >> 5;
  const int b    = blockIdx.x;
  const int l16  = lane & 15;
  const int lhi  = lane >> 4;

  // ---- phase 1: stage x1, x2 -> LDS as f16 A-fragments -------------------
  {
    const float* src1 = gx1 + (size_t)b * (NTOK * DIM);
    const float* src2 = gx2 + (size_t)b * (NTOK * DIM);
    #pragma unroll
    for (int i = 0; i < 8; ++i) {
      int gidx = tid + (i << 8);            // (token m, channel-group cg of 8)
      int m = gidx >> 5, cg = gidx & 31;
      int qq = cg & 3;
      int g  = qq & 1;
      int e0 = (qq >> 1) << 3;
      int laneD = (m & 15) + (g << 4);
      int frag  = ((m >> 4) << 3) + (cg >> 2);
      int dsto  = (frag * 32 + laneD) * 16 + e0;
      const float* p1 = src1 + m * DIM + (cg << 3);
      const float* p2 = src2 + m * DIM + (cg << 3);
      v4f a0 = *(const v4f*)p1; v4f a1 = *(const v4f*)(p1 + 4);
      v4f c0 = *(const v4f*)p2; v4f c1 = *(const v4f*)(p2 + 4);
      v8h h1, h2;
      #pragma unroll
      for (int j = 0; j < 4; ++j) {
        h1[j] = (_Float16)a0[j]; h1[j + 4] = (_Float16)a1[j];
        h2[j] = (_Float16)c0[j]; h2[j + 4] = (_Float16)c1[j];
      }
      *(v8h*)(smX1 + dsto) = h1;
      *(v8h*)(smX2 + dsto) = h2;
    }
  }
  __syncthreads();

  // ---- phase 2: q = x1 @ Wq ; [k|v] = x2 @ Wkv ---------------------------
  #pragma unroll 1
  for (int i = 0; i < 8; ++i) {               // 64 q-tiles / 8 waves
    int tt = wave * 8 + i;
    int mT = tt >> 4, nT = tt & 15;
    v8f acc = {};
    #pragma unroll
    for (int kt = 0; kt < 8; ++kt)
      acc = wmma16(ld_frag_lds(smX1, mT * 8 + kt, lane),
                   ld_frag_glb(gWq, kt * 16 + nT, lane), acc);
    int c = nT * 16 + l16, h = c >> 5, d = c & 31;
    int g = (d >> 3) & 1, e = ((d >> 4) << 3) + (d & 7);
    #pragma unroll
    for (int r = 0; r < 8; ++r) {             // scatter to q A-frag (K = d)
      int m = mT * 16 + r + (lhi << 3);
      smQ[((h * 4 + mT) * 32 + ((m & 15) + (g << 4))) * 16 + e] = (_Float16)acc[r];
    }
  }
  #pragma unroll 1
  for (int i = 0; i < 16; ++i) {              // 128 kv-tiles / 8 waves
    int tt = wave * 16 + i;
    int mT = tt >> 5, nT = tt & 31;
    v8f acc = {};
    #pragma unroll
    for (int kt = 0; kt < 8; ++kt)
      acc = wmma16(ld_frag_lds(smX2, mT * 8 + kt, lane),
                   ld_frag_glb(gWkv, kt * 32 + nT, lane), acc);
    int c = nT * 16 + l16;
    if (c < 256) {                            // K: B-frag of k^T (K=d, N=token)
      int h = c >> 5, d = c & 31;
      int g = (d >> 3) & 1, e = ((d >> 4) << 3) + (d & 7);
      #pragma unroll
      for (int r = 0; r < 8; ++r) {
        int m = mT * 16 + r + (lhi << 3);
        smK[((h * 4 + mT) * 32 + ((m & 15) + (g << 4))) * 16 + e] = (_Float16)acc[r];
      }
    } else {                                  // V: B-frag (K=token, N=d)
      int cv = c - 256, h = cv >> 5, d = cv & 31;
      #pragma unroll
      for (int r = 0; r < 8; ++r) {
        int m = mT * 16 + r + (lhi << 3);
        int kk = m & 31;
        int g = (kk >> 3) & 1, e = ((kk >> 4) << 3) + (kk & 7);
        smV[(((h * 2 + (m >> 5)) * 2 + (d >> 4)) * 32 + ((d & 15) + (g << 4))) * 16 + e]
            = (_Float16)acc[r];
      }
    }
  }
  __syncthreads();

  // ---- phase 3: per-head attention (wave h), softmax, motion, attn@V -----
  {
    const int h = wave;
    _Float16* const stg = smStage + wave * 1024;   // 16 rows x 64 keys f16
    #pragma unroll 1
    for (int mT = 0; mT < 4; ++mT) {
      v16h aq = *(const v16h*)(smQ + ((h * 4 + mT) * 32 + lane) * 16);
      float P[4][8];
      #pragma unroll
      for (int kt = 0; kt < 4; ++kt) {             // S = q @ k^T  (K=32, 1 WMMA)
        v8f acc = {};
        acc = wmma16(aq, ld_frag_lds(smK, h * 4 + kt, lane), acc);
        #pragma unroll
        for (int r = 0; r < 8; ++r) P[kt][r] = acc[r] * ATT_SCALE;
      }
      // softmax over 64 keys: 4 tiles in-lane + 16-lane shuffle reduce
      float mx[8], sum[8];
      #pragma unroll
      for (int r = 0; r < 8; ++r)
        mx[r] = fmaxf(fmaxf(P[0][r], P[1][r]), fmaxf(P[2][r], P[3][r]));
      #pragma unroll
      for (int s = 1; s < 16; s <<= 1) {
        #pragma unroll
        for (int r = 0; r < 8; ++r) mx[r] = fmaxf(mx[r], __shfl_xor(mx[r], s, 32));
      }
      #pragma unroll
      for (int r = 0; r < 8; ++r) {
        #pragma unroll
        for (int kt = 0; kt < 4; ++kt) P[kt][r] = __expf(P[kt][r] - mx[r]);
        sum[r] = P[0][r] + P[1][r] + P[2][r] + P[3][r];
      }
      #pragma unroll
      for (int s = 1; s < 16; s <<= 1) {
        #pragma unroll
        for (int r = 0; r < 8; ++r) sum[r] += __shfl_xor(sum[r], s, 32);
      }
      #pragma unroll
      for (int r = 0; r < 8; ++r) {
        float inv = 1.0f / sum[r];
        #pragma unroll
        for (int kt = 0; kt < 4; ++kt) P[kt][r] *= inv;
      }
      // motion: E[col_key]-col_q, E[row_key]-row_q  (rows sum to 1)
      {
        float colk = (float)(l16 & 7);
        float rb   = (float)(l16 >> 3);
        float mxp[8], myp[8];
        #pragma unroll
        for (int r = 0; r < 8; ++r) {
          float ps = P[0][r] + P[1][r] + P[2][r] + P[3][r];
          mxp[r] = ps * colk;   // key%8 == l16%8 for every tile
          myp[r] = 2.0f * P[1][r] + 4.0f * P[2][r] + 6.0f * P[3][r] + ps * rb;
        }
        #pragma unroll
        for (int s = 1; s < 16; s <<= 1) {
          #pragma unroll
          for (int r = 0; r < 8; ++r) {
            mxp[r] += __shfl_xor(mxp[r], s, 32);
            myp[r] += __shfl_xor(myp[r], s, 32);
          }
        }
        if (l16 == 0) {
          #pragma unroll
          for (int r = 0; r < 8; ++r) {
            int p = mT * 16 + r + (lhi << 3);
            smMotion[h * 64 + p]       = mxp[r] - (float)(p & 7);
            smMotion[512 + h * 64 + p] = myp[r] - (float)(p >> 3);
          }
        }
      }
      // re-layout P: C-frag -> A-frag via per-wave LDS bounce
      #pragma unroll
      for (int kt = 0; kt < 4; ++kt) {
        #pragma unroll
        for (int r = 0; r < 8; ++r)
          stg[(r + (lhi << 3)) * 64 + kt * 16 + l16] = (_Float16)P[kt][r];
      }
      v16h pa[2];
      #pragma unroll
      for (int k2 = 0; k2 < 2; ++k2) {
        V16U u;
        u.h[0] = *(const v8h*)(stg + l16 * 64 + k2 * 32 + (lhi << 3));
        u.h[1] = *(const v8h*)(stg + l16 * 64 + k2 * 32 + 16 + (lhi << 3));
        pa[k2] = u.v;
      }
      // attn @ v  -> xattn A-frags (channel tile of 32 == head h)
      #pragma unroll
      for (int nt = 0; nt < 2; ++nt) {
        v8f acc = {};
        #pragma unroll
        for (int k2 = 0; k2 < 2; ++k2)
          acc = wmma16(pa[k2],
                       *(const v16h*)(smV + (((h * 2 + k2) * 2 + nt) * 32 + lane) * 16),
                       acc);
        int d = nt * 16 + l16;
        int g = (d >> 3) & 1, e = ((d >> 4) << 3) + (d & 7);
        #pragma unroll
        for (int r = 0; r < 8; ++r) {
          int m = mT * 16 + r + (lhi << 3);
          smX1[((mT * 8 + h) * 32 + ((m & 15) + (g << 4))) * 16 + e] = (_Float16)acc[r];
        }
      }
    }
  }
  __syncthreads();

  // ---- phase 4: out = xattn @ Wproj + bproj ------------------------------
  #pragma unroll 1
  for (int i = 0; i < 8; ++i) {
    int tt = wave * 8 + i;
    int mT = tt >> 4, nT = tt & 15;
    v8f acc = {};
    #pragma unroll
    for (int kt = 0; kt < 8; ++kt)
      acc = wmma16(ld_frag_lds(smX1, mT * 8 + kt, lane),
                   ld_frag_glb(gWproj, kt * 16 + nT, lane), acc);
    int n = nT * 16 + l16;
    float bias = bproj[n];
    #pragma unroll
    for (int r = 0; r < 8; ++r) {
      int m = mT * 16 + r + (lhi << 3);
      out[(size_t)b * (NTOK * DIM) + m * DIM + n] = acc[r] + bias;
    }
  }

  // ---- motion MLP: (2B,64,8) -> gelu -> (2B,64,1) -> (B,64,2) ------------
  if (tid < 128) {
    int c = tid >> 6, p = tid & 63;
    float hsum[4];
    #pragma unroll
    for (int j = 0; j < 4; ++j) hsum[j] = bm1[j];
    #pragma unroll
    for (int hh = 0; hh < 8; ++hh) {
      float mv = smMotion[c * 512 + hh * 64 + p];
      #pragma unroll
      for (int j = 0; j < 4; ++j) hsum[j] += mv * Wm1[hh * 4 + j];
    }
    float o = bm2[0];
    #pragma unroll
    for (int j = 0; j < 4; ++j) {
      float gl = 0.5f * hsum[j] * (1.0f + erff(hsum[j] * 0.70710678118654752f));
      o += gl * Wm2[j];
    }
    out[(size_t)nB * (NTOK * DIM) + ((size_t)b * NTOK + p) * 2 + c] = o;
  }
}

// -------------------------------------------------------------------------
extern "C" void kernel_launch(void* const* d_in, const int* in_sizes, int n_in,
                              void* d_out, int out_size, void* d_ws, size_t ws_size,
                              hipStream_t stream) {
  const float* x1    = (const float*)d_in[0];
  const float* x2    = (const float*)d_in[1];
  // d_in[2]=H, d_in[3]=W unused
  const float* Wq    = (const float*)d_in[4];
  const float* Wkv   = (const float*)d_in[5];
  const float* Wproj = (const float*)d_in[6];
  const float* bproj = (const float*)d_in[7];
  const float* Wm1   = (const float*)d_in[8];
  const float* bm1   = (const float*)d_in[9];
  const float* Wm2   = (const float*)d_in[10];
  const float* bm2   = (const float*)d_in[11];

  int nB = in_sizes[0] / (NTOK * DIM);
  _Float16* ws = (_Float16*)d_ws;

  prep_weights_kernel<<<1024, 256, 0, stream>>>(Wq, Wkv, Wproj, ws);
  attn_motion_kernel<<<nB, 256, 0, stream>>>(x1, x2, ws, bproj,
                                             Wm1, bm1, Wm2, bm2,
                                             (float*)d_out, nB);
}
